// PointNetPP_22668837388950
// MI455X (gfx1250) — matrix operand order repaired and checked by reading, not compiled, not run on hardware
//
#include <hip/hip_runtime.h>
#include <hip/hip_bf16.h>

typedef __attribute__((ext_vector_type(16))) __bf16 v16bf;
typedef __attribute__((ext_vector_type(8)))  __bf16 v8bf;
typedef __attribute__((ext_vector_type(4)))  __bf16 v4bf;
typedef __attribute__((ext_vector_type(8)))  float  v8f;
typedef __attribute__((ext_vector_type(4)))  float  f4v;

#define LRELU 0.01f
#define BN_EPS 1e-5f

// ---------------------------------------------------------------------------
// Farthest point sampling: one block per batch, dists cached in LDS.
// ---------------------------------------------------------------------------
__global__ __launch_bounds__(256)
void fps_kernel(const float* __restrict__ pos, int* __restrict__ out,
                int Npts, int S) {
  __shared__ float dists[4096];
  __shared__ float rv[256];
  __shared__ int   ri[256];
  __shared__ int   s_last;
  int b = blockIdx.x;
  const float* P = pos + (size_t)b * Npts * 3;
  for (int i = threadIdx.x; i < Npts; i += 256) dists[i] = 1e10f;
  if (threadIdx.x == 0) { s_last = 0; out[b * S] = 0; }
  __syncthreads();
  for (int it = 1; it < S; ++it) {
    int last = s_last;
    float lx = P[last * 3 + 0], ly = P[last * 3 + 1], lz = P[last * 3 + 2];
    float best = -1.0f; int bidx = 0;
    for (int i = threadIdx.x; i < Npts; i += 256) {
      float dx = P[i * 3 + 0] - lx;
      float dy = P[i * 3 + 1] - ly;
      float dz = P[i * 3 + 2] - lz;
      float d  = dx * dx + dy * dy + dz * dz;
      float dm = fminf(dists[i], d);
      dists[i] = dm;
      if (dm > best) { best = dm; bidx = i; }
    }
    rv[threadIdx.x] = best; ri[threadIdx.x] = bidx;
    __syncthreads();
    for (int s = 128; s > 0; s >>= 1) {
      if (threadIdx.x < s) {
        if (rv[threadIdx.x + s] > rv[threadIdx.x]) {
          rv[threadIdx.x] = rv[threadIdx.x + s];
          ri[threadIdx.x] = ri[threadIdx.x + s];
        }
      }
      __syncthreads();
    }
    if (threadIdx.x == 0) { s_last = ri[0]; out[b * S + it] = ri[0]; }
    __syncthreads();
  }
}

__global__ void gather_pos_kernel(const float* __restrict__ pos,
                                  const int* __restrict__ fpsidx,
                                  float* __restrict__ newpos,
                                  int Npts, int S, int total) {
  int i = blockIdx.x * blockDim.x + threadIdx.x;  // over B*S
  if (i >= total) return;
  int b = i / S;
  int p = fpsidx[i];
  newpos[i * 3 + 0] = pos[((size_t)b * Npts + p) * 3 + 0];
  newpos[i * 3 + 1] = pos[((size_t)b * Npts + p) * 3 + 1];
  newpos[i * 3 + 2] = pos[((size_t)b * Npts + p) * 3 + 2];
}

// ---------------------------------------------------------------------------
// Ball query: one wave32 per query point; first-K-in-index-order via ballot.
// ---------------------------------------------------------------------------
__global__ __launch_bounds__(32)
void ball_query_kernel(const float* __restrict__ pos,
                       const float* __restrict__ newpos,
                       int* __restrict__ idx,
                       int Npts, int S, int Kk, float r2) {
  int q = blockIdx.x;            // b*S + s
  int b = q / S;
  int lane = threadIdx.x;
  const float* P = pos + (size_t)b * Npts * 3;
  float qx = newpos[q * 3 + 0], qy = newpos[q * 3 + 1], qz = newpos[q * 3 + 2];
  int* o = idx + (size_t)q * Kk;
  int cnt = 0;
  int first = -1;
  for (int base = 0; base < Npts && cnt < Kk; base += 32) {
    int p = base + lane;
    float dx = P[p * 3 + 0] - qx;
    float dy = P[p * 3 + 1] - qy;
    float dz = P[p * 3 + 2] - qz;
    bool in = (dx * dx + dy * dy + dz * dz) < r2;
    unsigned m32 = (unsigned)__ballot(in);
    if (first < 0 && m32) first = base + __builtin_ctz(m32);
    if (in) {
      unsigned below = (lane == 0) ? 0u : (m32 & ((1u << lane) - 1u));
      int ppos = cnt + __builtin_popcount(below);
      if (ppos < Kk) o[ppos] = p;
    }
    cnt += __builtin_popcount(m32);
  }
  if (cnt > Kk) cnt = Kk;
  int fill = (first >= 0) ? first : 0;
  for (int i2 = cnt + lane; i2 < Kk; i2 += 32) o[i2] = fill;
}

// ---------------------------------------------------------------------------
// Grouping: relative coordinates into a channel-PADDED feature row (pad = 0)
// so downstream GEMMs have K % 32 == 0 and need no A-side bounds checks.
// ---------------------------------------------------------------------------
__global__ void group_pos_kernel(const float* __restrict__ pos,
                                 const float* __restrict__ newpos,
                                 const int* __restrict__ idx,
                                 float* __restrict__ feat,   // [total][32]
                                 int Npts, int S, int Kk, int total) {
  int i = blockIdx.x * blockDim.x + threadIdx.x;  // over B*S*K
  if (i >= total) return;
  int s = (i / Kk) % S;
  int b = i / (Kk * S);
  int p = idx[i];
  int q = b * S + s;
  float* dst = feat + (size_t)i * 32;
  dst[0] = pos[((size_t)b * Npts + p) * 3 + 0] - newpos[q * 3 + 0];
  dst[1] = pos[((size_t)b * Npts + p) * 3 + 1] - newpos[q * 3 + 1];
  dst[2] = pos[((size_t)b * Npts + p) * 3 + 2] - newpos[q * 3 + 2];
  for (int c = 3; c < 32; ++c) dst[c] = 0.0f;
}

__global__ void group_feat_kernel(const float* __restrict__ pos,
                                  const float* __restrict__ newpos,
                                  const float* __restrict__ f1,
                                  const int* __restrict__ idx,
                                  float* __restrict__ feat,  // [total][224]
                                  int Npts, int S, int Kk, int Cf, int total) {
  int i = blockIdx.x * blockDim.x + threadIdx.x;  // over B*S*K
  if (i >= total) return;
  int s = (i / Kk) % S;
  int b = i / (Kk * S);
  int p = idx[i];
  int q = b * S + s;
  float* dst = feat + (size_t)i * 224;
  dst[0] = pos[((size_t)b * Npts + p) * 3 + 0] - newpos[q * 3 + 0];
  dst[1] = pos[((size_t)b * Npts + p) * 3 + 1] - newpos[q * 3 + 1];
  dst[2] = pos[((size_t)b * Npts + p) * 3 + 2] - newpos[q * 3 + 2];
  const float* src = f1 + ((size_t)b * Npts + p) * Cf;
  for (int c = 0; c < Cf; ++c) dst[3 + c] = src[c];
  for (int c = 3 + Cf; c < 224; ++c) dst[c] = 0.0f;
}

// ---------------------------------------------------------------------------
// Tiled WMMA GEMM: C[rows x N] = A[rows x Kpad] * W[Kw x N], bf16 / f32 acc.
// Preconditions: rows % 128 == 0, Kpad % 32 == 0, N == NSUB*16*gridDim.y,
//                A columns >= Kw are zero.
// Block = 256 threads = 8 waves; tile 128(M) x NSUB*16(N).
// A tiles stream global->LDS with GLOBAL_LOAD_ASYNC_TO_LDS_B128 (ASYNCcnt),
// double-buffered so tile k+1 copies while tile k feeds the WMMAs.
// ---------------------------------------------------------------------------
template <int NSUB>
__global__ __launch_bounds__(256)
void gemm_bf16_wmma(const float* __restrict__ A, const float* __restrict__ W,
                    float* __restrict__ C, int rows, int Kpad, int Kw, int N) {
  __shared__ float  sA32[2][128][32];     // 32 KB double-buffered f32 A tiles
  __shared__ __bf16 sBT[NSUB * 16][32];   // transposed bf16 (n, k)
  const int tileM = blockIdx.x * 128;
  const int tileN = blockIdx.y * (NSUB * 16);
  const int t = threadIdx.x;
  const int wave = t >> 5, lane = t & 31;
  const int m16 = lane & 15, h = lane >> 4;

  // Issue async global->LDS copy of the 128x32 f32 A tile at column k0.
  auto issueA = [&](int k0, int buf) {
#pragma unroll
    for (int ps = 0; ps < 4; ++ps) {
      int idx = ps * 256 + t;              // float4 index in tile
      int r = idx >> 3;
      int c = (idx & 7) * 4;
      unsigned ldsa = (unsigned)(uintptr_t)(&sA32[buf][r][c]);
      const float* g = A + (size_t)(tileM + r) * Kpad + (k0 + c);
      asm volatile("global_load_async_to_lds_b128 %0, %1, off"
                   :: "v"(ldsa), "v"(g) : "memory");
    }
  };

  v8f acc[NSUB];
#pragma unroll
  for (int j = 0; j < NSUB; ++j) acc[j] = v8f{};

  issueA(0, 0);

  for (int k0 = 0; k0 < Kpad; k0 += 32) {
    const int  cur  = (k0 >> 5) & 1;
    const bool more = (k0 + 32) < Kpad;
    if (more) issueA(k0 + 32, cur ^ 1);   // prefetch next tile into other buf

    // ---- stage B transposed while async copies are in flight ----
    // clamp-and-select instead of branch-around-load for the K-tail guard
#pragma unroll
    for (int ps = 0; ps < NSUB * 2; ++ps) {
      int idx = ps * 256 + t;              // element index
      int n = idx >> 5;
      int k = idx & 31;
      int gk = k0 + k;
      int gkc = (gk < Kw) ? gk : (Kw - 1);
      float v = W[(size_t)gkc * N + (tileN + n)];
      sBT[n][k] = (__bf16)((gk < Kw) ? v : 0.0f);
    }
    // current tile's 4 async copies done (in-order); next tile may be in flight
    if (more) asm volatile("s_wait_asynccnt 0x4" ::: "memory");
    else      asm volatile("s_wait_asynccnt 0x0" ::: "memory");
    __syncthreads();

    // ---- A fragment: four b128 LDS loads (f32) + packed bf16 converts ----
    const float* ar = &sA32[cur][wave * 16 + m16][0];
    f4v a0 = *(const f4v*)(ar + h * 8);
    f4v a1 = *(const f4v*)(ar + h * 8 + 4);
    f4v a2 = *(const f4v*)(ar + 16 + h * 8);
    f4v a3 = *(const f4v*)(ar + 16 + h * 8 + 4);
    v16bf af;
#pragma unroll
    for (int i = 0; i < 4; ++i) {
      af[i]      = (__bf16)a0[i];
      af[4 + i]  = (__bf16)a1[i];
      af[8 + i]  = (__bf16)a2[i];
      af[12 + i] = (__bf16)a3[i];
    }

    // ---- NSUB N-subtiles: B fragment = two b128 LDS loads each ----
#pragma unroll
    for (int j = 0; j < NSUB; ++j) {
      int n = j * 16 + m16;
      v8bf b0 = *(const v8bf*)(&sBT[n][h * 8]);
      v8bf b1 = *(const v8bf*)(&sBT[n][16 + h * 8]);
      v16bf bfr;
#pragma unroll
      for (int i = 0; i < 8; ++i) { bfr[i] = b0[i]; bfr[8 + i] = b1[i]; }
      acc[j] = __builtin_amdgcn_wmma_f32_16x16x32_bf16(
          false, af, false, bfr, (short)0, acc[j], false, false);
    }
    __syncthreads();
  }

#pragma unroll
  for (int j = 0; j < NSUB; ++j) {
#pragma unroll
    for (int r = 0; r < 8; ++r) {
      int row = tileM + wave * 16 + r + h * 8;
      int col = tileN + j * 16 + m16;
      C[(size_t)row * N + col] = acc[j][r];
    }
  }
}

// ---------------------------------------------------------------------------
// Per-column sum / sum-of-squares (one block per output channel, no atomics).
// ---------------------------------------------------------------------------
__global__ __launch_bounds__(256)
void colstats_kernel(const float* __restrict__ Y, float* __restrict__ stats,
                     int rows, int N) {
  __shared__ float s1[256], s2[256];
  int c = blockIdx.x;
  float a = 0.0f, b = 0.0f;
  for (int r = threadIdx.x; r < rows; r += 256) {
    float v = Y[(size_t)r * N + c];
    a += v; b += v * v;
  }
  s1[threadIdx.x] = a; s2[threadIdx.x] = b;
  __syncthreads();
  for (int s = 128; s > 0; s >>= 1) {
    if (threadIdx.x < s) {
      s1[threadIdx.x] += s1[threadIdx.x + s];
      s2[threadIdx.x] += s2[threadIdx.x + s];
    }
    __syncthreads();
  }
  if (threadIdx.x == 0) { stats[c] = s1[0]; stats[N + c] = s2[0]; }
}

__global__ void bn_lrelu_kernel(float* __restrict__ Y,
                                const float* __restrict__ stats,
                                const float* __restrict__ g,
                                const float* __restrict__ bb,
                                int rows, int N, size_t total) {
  size_t i = (size_t)blockIdx.x * blockDim.x + threadIdx.x;
  if (i >= total) return;
  int c = (int)(i % N);
  float inv = 1.0f / (float)rows;
  float mean = stats[c] * inv;
  float var  = stats[N + c] * inv - mean * mean;
  float y = g[c] * (Y[i] - mean) * rsqrtf(var + BN_EPS) + bb[c];
  Y[i] = (y >= 0.0f) ? y : LRELU * y;
}

// ---------------------------------------------------------------------------
// Max over neighbor dimension K, writing into a channel-concat destination.
// ---------------------------------------------------------------------------
__global__ void maxk_kernel(const float* __restrict__ Y, float* __restrict__ out,
                            int BS, int Kk, int C, int outStride, int outOff) {
  int i = blockIdx.x * blockDim.x + threadIdx.x;  // over BS*C
  if (i >= BS * C) return;
  int bs = i / C, c = i % C;
  float m = -3.0e38f;
  for (int k = 0; k < Kk; ++k)
    m = fmaxf(m, Y[((size_t)bs * Kk + k) * C + c]);
  out[(size_t)bs * outStride + outOff + c] = m;
}

__global__ void smax_kernel(const float* __restrict__ f, float* __restrict__ out,
                            int S, int C, int total) {
  int i = blockIdx.x * blockDim.x + threadIdx.x;  // over B*C
  if (i >= total) return;
  int b = i / C, c = i % C;
  float m = -3.0e38f;
  for (int s = 0; s < S; ++s) m = fmaxf(m, f[((size_t)b * S + s) * C + c]);
  out[i] = m;
}

// ---------------------------------------------------------------------------
// FC layers: X is exactly 16 rows -> one WMMA M-tile. One wave per 16 cols.
// K is always a multiple of 32 here, so no K guards; N guard is clamp+select.
// ---------------------------------------------------------------------------
__global__ __launch_bounds__(32)
void fc_wmma_kernel(const float* __restrict__ X, const float* __restrict__ W,
                    const float* __restrict__ bias, float* __restrict__ out,
                    int K, int N, int relu) {
  int tileN = blockIdx.x * 16;
  int lane = threadIdx.x;
  int m16 = lane & 15;
  int h   = lane >> 4;
  int n   = tileN + m16;
  int nc  = (n < N) ? n : (N - 1);
  v8f acc = {};
  for (int k0 = 0; k0 < K; k0 += 32) {
    v16bf af, bf;
#pragma unroll
    for (int v = 0; v < 8; ++v) {
      int kbA = (v < 4) ? (v * 2 + h * 8) : (16 + (v - 4) * 2 + h * 8);
      int kbB = (v < 4) ? (v * 2 + h * 8) : (16 + (v - 4) * 2 + h * 8);
#pragma unroll
      for (int j = 0; j < 2; ++j) {
        float av = X[(size_t)m16 * K + (k0 + kbA + j)];
        float bv = W[(size_t)(k0 + kbB + j) * N + nc];
        af[2 * v + j] = (__bf16)av;
        bf[2 * v + j] = (__bf16)((n < N) ? bv : 0.0f);
      }
    }
    acc = __builtin_amdgcn_wmma_f32_16x16x32_bf16(false, af, false, bf,
                                                  (short)0, acc, false, false);
  }
#pragma unroll
  for (int r = 0; r < 8; ++r) {
    int row = r + h * 8;
    if (n < N) {
      float y = acc[r] + bias[n];
      if (relu) y = fmaxf(y, 0.0f);
      out[(size_t)row * N + n] = y;
    }
  }
}

__global__ void logsoftmax_kernel(const float* __restrict__ logits,
                                  float* __restrict__ out, int Bn, int C) {
  int b = threadIdx.x;
  if (b >= Bn) return;
  float mx = -3.0e38f;
  for (int c = 0; c < C; ++c) mx = fmaxf(mx, logits[b * C + c]);
  float s = 0.0f;
  for (int c = 0; c < C; ++c) s += expf(logits[b * C + c] - mx);
  float ls = logf(s);
  for (int c = 0; c < C; ++c) out[b * C + c] = logits[b * C + c] - mx - ls;
}

// ---------------------------------------------------------------------------
// Host orchestration
// ---------------------------------------------------------------------------
static void run_mlp_layer(const float* X, const float* W, const float* g,
                          const float* bb, float* Y, float* stats,
                          int rows, int Kpad, int Kw, int Cout,
                          hipStream_t stream) {
  if (Cout % 64 == 0) {
    dim3 gg(rows / 128, Cout / 64);
    gemm_bf16_wmma<4><<<gg, 256, 0, stream>>>(X, W, Y, rows, Kpad, Kw, Cout);
  } else {  // Cout == 32
    dim3 gg(rows / 128, 1);
    gemm_bf16_wmma<2><<<gg, 256, 0, stream>>>(X, W, Y, rows, Kpad, Kw, Cout);
  }
  colstats_kernel<<<Cout, 256, 0, stream>>>(Y, stats, rows, Cout);
  size_t tot = (size_t)rows * Cout;
  bn_lrelu_kernel<<<(unsigned)((tot + 255) / 256), 256, 0, stream>>>(
      Y, stats, g, bb, rows, Cout, tot);
}

extern "C" void kernel_launch(void* const* d_in, const int* in_sizes, int n_in,
                              void* d_out, int out_size, void* d_ws, size_t ws_size,
                              hipStream_t stream) {
  (void)in_sizes; (void)n_in; (void)out_size; (void)ws_size;

  const int B = 16, N0 = 4096;
  const int S1 = 512, S2 = 128;

  // ---- parameter pointers (setup_inputs dict order) ----
  const float* x = (const float*)d_in[0];
  const float *saW[4][3], *saG[4][3], *saB[4][3];
  int p = 1;
  for (int sc = 0; sc < 4; ++sc)
    for (int l = 0; l < 3; ++l) {
      saW[sc][l] = (const float*)d_in[p++];
      saG[sc][l] = (const float*)d_in[p++];
      saB[sc][l] = (const float*)d_in[p++];
    }
  const float *fcW[3], *fcB[3];
  for (int i = 0; i < 3; ++i) {
    fcW[i] = (const float*)d_in[p++];
    fcB[i] = (const float*)d_in[p++];
  }

  // ---- workspace carve ----
  char* ws = (char*)d_ws;
  size_t off = 0;
  auto carve = [&](size_t bytes) {
    void* q = ws + off;
    off = (off + bytes + 255) & ~(size_t)255;
    return q;
  };
  int*   fpsidx1 = (int*)  carve((size_t)B * S1 * 4);
  float* newpos1 = (float*)carve((size_t)B * S1 * 3 * 4);
  int*   fpsidx2 = (int*)  carve((size_t)B * S2 * 4);
  float* newpos2 = (float*)carve((size_t)B * S2 * 3 * 4);
  int*   idx1a   = (int*)  carve((size_t)B * S1 * 16 * 4);
  int*   idx1b   = (int*)  carve((size_t)B * S1 * 32 * 4);
  int*   idx2a   = (int*)  carve((size_t)B * S2 * 32 * 4);
  int*   idx2b   = (int*)  carve((size_t)B * S2 * 64 * 4);
  float* stats   = (float*)carve(1024 * 4);
  float* f1      = (float*)carve((size_t)B * S1 * 192 * 4);
  float* f2      = (float*)carve((size_t)B * S2 * 768 * 4);
  float* gbuf    = (float*)carve((size_t)B * 768 * 4);
  float* h1      = (float*)carve((size_t)B * 256 * 4);
  float* h2      = (float*)carve((size_t)B * 128 * 4);
  float* logits  = (float*)carve((size_t)B * 40 * 4);
  float* BUF0    = (float*)carve((size_t)33554432 * 4);  // 128 MB ping
  float* BUF1    = (float*)carve((size_t)67108864 * 4);  // 256 MB pong

  // ================= SA1 =================
  fps_kernel<<<B, 256, 0, stream>>>(x, fpsidx1, N0, S1);
  gather_pos_kernel<<<(B * S1 + 255) / 256, 256, 0, stream>>>(
      x, fpsidx1, newpos1, N0, S1, B * S1);

  // scale 0: r=0.1, K=16, mlp 3->32->32->64 (input padded to 32)
  {
    int K = 16, rows = B * S1 * K;
    ball_query_kernel<<<B * S1, 32, 0, stream>>>(x, newpos1, idx1a, N0, S1, K, 0.01f);
    group_pos_kernel<<<(rows + 255) / 256, 256, 0, stream>>>(
        x, newpos1, idx1a, BUF0, N0, S1, K, rows);
    run_mlp_layer(BUF0, saW[0][0], saG[0][0], saB[0][0], BUF1, stats, rows, 32, 3, 32, stream);
    run_mlp_layer(BUF1, saW[0][1], saG[0][1], saB[0][1], BUF0, stats, rows, 32, 32, 32, stream);
    run_mlp_layer(BUF0, saW[0][2], saG[0][2], saB[0][2], BUF1, stats, rows, 32, 32, 64, stream);
    maxk_kernel<<<(B * S1 * 64 + 255) / 256, 256, 0, stream>>>(
        BUF1, f1, B * S1, K, 64, 192, 0);
  }
  // scale 1: r=0.2, K=32, mlp 3->64->64->128 (input padded to 32)
  {
    int K = 32, rows = B * S1 * K;
    ball_query_kernel<<<B * S1, 32, 0, stream>>>(x, newpos1, idx1b, N0, S1, K, 0.04f);
    group_pos_kernel<<<(rows + 255) / 256, 256, 0, stream>>>(
        x, newpos1, idx1b, BUF0, N0, S1, K, rows);
    run_mlp_layer(BUF0, saW[1][0], saG[1][0], saB[1][0], BUF1, stats, rows, 32, 3, 64, stream);
    run_mlp_layer(BUF1, saW[1][1], saG[1][1], saB[1][1], BUF0, stats, rows, 64, 64, 64, stream);
    run_mlp_layer(BUF0, saW[1][2], saG[1][2], saB[1][2], BUF1, stats, rows, 64, 64, 128, stream);
    maxk_kernel<<<(B * S1 * 128 + 255) / 256, 256, 0, stream>>>(
        BUF1, f1, B * S1, K, 128, 192, 64);
  }

  // ================= SA2 =================
  fps_kernel<<<B, 256, 0, stream>>>(newpos1, fpsidx2, S1, S2);
  gather_pos_kernel<<<(B * S2 + 255) / 256, 256, 0, stream>>>(
      newpos1, fpsidx2, newpos2, S1, S2, B * S2);

  // scale 0: r=0.2, K=32, mlp 195->128->128->256 (input padded to 224)
  {
    int K = 32, rows = B * S2 * K;
    ball_query_kernel<<<B * S2, 32, 0, stream>>>(newpos1, newpos2, idx2a, S1, S2, K, 0.04f);
    group_feat_kernel<<<(rows + 255) / 256, 256, 0, stream>>>(
        newpos1, newpos2, f1, idx2a, BUF0, S1, S2, K, 192, rows);
    run_mlp_layer(BUF0, saW[2][0], saG[2][0], saB[2][0], BUF1, stats, rows, 224, 195, 128, stream);
    run_mlp_layer(BUF1, saW[2][1], saG[2][1], saB[2][1], BUF0, stats, rows, 128, 128, 128, stream);
    run_mlp_layer(BUF0, saW[2][2], saG[2][2], saB[2][2], BUF1, stats, rows, 128, 128, 256, stream);
    maxk_kernel<<<(B * S2 * 256 + 255) / 256, 256, 0, stream>>>(
        BUF1, f2, B * S2, K, 256, 768, 0);
  }
  // scale 1: r=0.4, K=64, mlp 195->256->256->512 (input padded to 224)
  {
    int K = 64, rows = B * S2 * K;
    ball_query_kernel<<<B * S2, 32, 0, stream>>>(newpos1, newpos2, idx2b, S1, S2, K, 0.16f);
    group_feat_kernel<<<(rows + 255) / 256, 256, 0, stream>>>(
        newpos1, newpos2, f1, idx2b, BUF0, S1, S2, K, 192, rows);
    run_mlp_layer(BUF0, saW[3][0], saG[3][0], saB[3][0], BUF1, stats, rows, 224, 195, 256, stream);
    run_mlp_layer(BUF1, saW[3][1], saG[3][1], saB[3][1], BUF0, stats, rows, 256, 256, 256, stream);
    run_mlp_layer(BUF0, saW[3][2], saG[3][2], saB[3][2], BUF1, stats, rows, 256, 256, 512, stream);
    maxk_kernel<<<(B * S2 * 512 + 255) / 256, 256, 0, stream>>>(
        BUF1, f2, B * S2, K, 512, 768, 256);
  }

  // ================= Head =================
  smax_kernel<<<(B * 768 + 255) / 256, 256, 0, stream>>>(f2, gbuf, S2, 768, B * 768);
  fc_wmma_kernel<<<256 / 16, 32, 0, stream>>>(gbuf, fcW[0], fcB[0], h1, 768, 256, 1);
  fc_wmma_kernel<<<128 / 16, 32, 0, stream>>>(h1,   fcW[1], fcB[1], h2, 256, 128, 1);
  fc_wmma_kernel<<<(40 + 15) / 16, 32, 0, stream>>>(h2, fcW[2], fcB[2], logits, 128, 40, 0);
  logsoftmax_kernel<<<1, 32, 0, stream>>>(logits, (float*)d_out, B, 40);
}
